// GyroplaneConvLayer_61177514164690
// MI455X (gfx1250) — compile-verified, matrix operand order checked by reading, not run on hardware
//
#include <hip/hip_runtime.h>
#include <cmath>

typedef _Float16     v16h  __attribute__((ext_vector_type(16)));
typedef _Float16     v8h   __attribute__((ext_vector_type(8)));
typedef float        v8f   __attribute__((ext_vector_type(8)));
typedef unsigned int u32x4 __attribute__((ext_vector_type(4)));
typedef int          i32x4 __attribute__((ext_vector_type(4)));
typedef int          i32x8 __attribute__((ext_vector_type(8)));

#if defined(__has_builtin)
#if __has_builtin(__builtin_amdgcn_tensor_load_to_lds)
#define GYRO_HAVE_TDM 1
#endif
#endif
#if defined(__has_include)
#if __has_include(<hip/amd_detail/amd_gfx1250_TDM.h>)
#define GYRO_TDM_6ARG 1
#endif
#endif

namespace {
constexpr int    kB         = 16;
constexpr int    kD         = 64;
constexpr int    kO         = 128;
constexpr int    kN         = 30;
constexpr int    kNP        = 32;                    // padded edge (30 + 2)
constexpr int    kRows      = kB * kNP * kNP * kNP;  // 524288 padded points
constexpr size_t kDistElems = (size_t)kRows * kO;    // 67,108,864 floats
constexpr float  kMinNorm   = 1e-15f;
constexpr float  kMaxNorm   = 1.0f - 4e-3f;          // geoopt projection radius
constexpr int    kLdsStride = 80;                    // halfs/row, 16B-aligned runs
constexpr int    kBMatHalfs = 4 * kO * kD;           // pHi|pLo|aHi|aLo = 32768 halfs
}

// ---------------------------------------------------------------------------
// Kernel 1: gyroplane parameters. p = coefP*w and a = betaC*w, so per-channel
// scalars reduce to closed forms of w2 = |w|^2. One thread per channel o.
// ---------------------------------------------------------------------------
__global__ void gyro_prep(const float* __restrict__ w, const float* __restrict__ bias,
                          _Float16* __restrict__ bmat,   // [pHi|pLo|aHi|aLo]
                          float* __restrict__ p2s, float* __restrict__ pas,
                          float* __restrict__ ans) {
  const int o = threadIdx.x;
  if (o >= kO) return;
  _Float16* pHi = bmat;
  _Float16* pLo = bmat + kO * kD;
  _Float16* aHi = bmat + 2 * kO * kD;
  _Float16* aLo = bmat + 3 * kO * kD;
  const float* wr = w + o * kD;
  float w2 = 0.f;
  for (int d = 0; d < kD; ++d) w2 += wr[d] * wr[d];
  const float b0  = bias[o];
  const float sqw = sqrtf(w2);
  // expmap0(u = w*b): gamma = tanh(clip(|u|,15)) * u / |u|
  const float un = fmaxf(fabsf(b0) * sqw, kMinNorm);
  const float g  = tanhf(fminf(un, 15.f)) / un;
  float coefP = b0 * g;                         // p = coefP * w
  const float pn = fmaxf(fabsf(coefP) * sqw, kMinNorm);
  if (pn > kMaxNorm) coefP *= kMaxNorm / pn;    // projx
  const float p2    = coefP * coefP * w2;
  const float betaC = fmaxf(1.f - p2, kMinNorm);     // a = betaC * w
  const float pa    = coefP * betaC * w2;
  const float an    = fmaxf(betaC * sqw, kMinNorm);
  p2s[o] = p2; pas[o] = pa; ans[o] = an;
  for (int d = 0; d < kD; ++d) {
    const float pf = coefP * wr[d];
    const float af = betaC * wr[d];
    const _Float16 ph = (_Float16)pf;
    const _Float16 ah = (_Float16)af;
    pHi[o * kD + d] = ph;
    pLo[o * kD + d] = (_Float16)(pf - (float)ph);
    aHi[o * kD + d] = ah;
    aLo[o * kD + d] = (_Float16)(af - (float)ah);
  }
}

// ---------------------------------------------------------------------------
// WMMA fragment builder (16-bit 16x32 operand layout, ISA 7.12.2):
//   lanes 0-15  : halfs 0..7 = K klo+0..7, halfs 8..15 = K klo+16..23, klo=0
//   lanes 16-31 : same with klo = 8
// Each run is 8 contiguous halfs = one 16-byte ds_load_b128.
// ---------------------------------------------------------------------------
__device__ __forceinline__ v16h frag_at(const _Float16* base, int halfOfs) {
  union { v16h v; v8h h[2]; } u;
  const _Float16* r = base + halfOfs;
  u.h[0] = *(const v8h*)(r);
  u.h[1] = *(const v8h*)(r + 16);
  return u.v;
}

__device__ __forceinline__ v8f wmma_f16(v16h a, v16h b, v8f c) {
  return __builtin_amdgcn_wmma_f32_16x16x32_f16(false, a, false, b,
                                                (short)0, c, false, false);
}

#if defined(GYRO_HAVE_TDM)
// TDM: DMA the contiguous 64KB B-operand block (4 x 128x64 f16) into LDS.
// D# per CDNA5 ISA: 2-D tensor, data_size=2B,
// tile 32768 x 1 elements, groups 2/3 unused (zero).
__device__ __forceinline__ void tdm_load_bmat(const _Float16* gsrc,
                                              unsigned ldsByteAddr) {
  const unsigned long long ga = (unsigned long long)(size_t)gsrc;
  const u32x4 g0 = {
      1u,                                            // count=1 (valid descriptor)
      ldsByteAddr,                                   // lds_addr
      (unsigned)(ga & 0xFFFFFFFFu),                  // global_addr[31:0]
      (unsigned)((ga >> 32) & 0x01FFFFFFu) | 0x80000000u  // ga[56:32] | type=2
  };
  const i32x8 g1 = {
      0x00010000,        // data_size=1 (2 bytes); no multicast/pad/iterate
      (int)0x80000000,   // tensor_dim0[15:0]=32768 in bits [63:48]
      0x00010000,        // tensor_dim1=1 in bits [95:80]
      (int)0x80000000,   // tile_dim0=32768 in bits [127:112]
      0x00000001,        // tile_dim1=1
      0x00008000,        // tensor_dim0_stride=32768 (low bits)
      0, 0
  };
  const i32x4 z4 = {0, 0, 0, 0};
#if defined(GYRO_TDM_6ARG)
  const i32x8 z8 = {0, 0, 0, 0, 0, 0, 0, 0};
  __builtin_amdgcn_tensor_load_to_lds(g0, g1, z4, z4, z8, 0);
#else
  __builtin_amdgcn_tensor_load_to_lds(g0, g1, z4, z4, 0);
#endif
}
#endif

// ---------------------------------------------------------------------------
// Kernel 2: fused hyperbolic-distance GEMM. One block = 32 padded points x all
// 128 channels. 8 waves: wave = (row tile in {0,1}) x (col pair in {0..3}).
// fp32 emulated via hi/lo f16 split: acc += Ah*Bh + Al*Bh + Ah*Bl.
// ---------------------------------------------------------------------------
__global__ __launch_bounds__(256) void gyro_dist(
    const float* __restrict__ x,
    const _Float16* __restrict__ bmat,      // [pHi|pLo|aHi|aLo], contiguous
    const float* __restrict__ p2s, const float* __restrict__ pas,
    const float* __restrict__ ans, float* __restrict__ dist) {
  __shared__ __align__(16) _Float16 sB[kBMatHalfs];     // 64KB B operands
  __shared__ __align__(16) _Float16 sHi[32 * kLdsStride];
  __shared__ __align__(16) _Float16 sLo[32 * kLdsStride];
  __shared__ float sPart[32 * 8];
  __shared__ float sX2[32];

  const int tid     = threadIdx.x;
  const int rowBase = blockIdx.x * 32;

#if defined(GYRO_HAVE_TDM)
  if (tid == 0) {
    tdm_load_bmat(bmat, (unsigned)(size_t)(void*)&sB[0]);
  }
#endif
  __builtin_prefetch(x + (size_t)blockIdx.x * 64, 0, 0);  // global_prefetch_b8

  // ---- Stage 32 rows x 64 dims of padded x into LDS as hi/lo f16 ----
  {
    const int r = tid >> 3, seg = tid & 7;        // 8 floats per thread
    const int m  = rowBase + r;
    const int b  = m >> 15;
    const int zp = (m >> 10) & 31;
    const int yp = (m >> 5) & 31;
    const int xq = m & 31;
    const bool border = (zp == 0) | (zp == 31) | (yp == 0) | (yp == 31) |
                        (xq == 0) | (xq == 31);
    float v[8];
    if (!border) {
      // x layout: [M = z*900 + y*30 + x][B][D]
      const size_t mi = (size_t)((zp - 1) * 900 + (yp - 1) * 30 + (xq - 1));
      const float* src = x + (mi * kB + b) * kD + seg * 8;
      const float4 v0 = ((const float4*)src)[0];
      const float4 v1 = ((const float4*)src)[1];
      v[0] = v0.x; v[1] = v0.y; v[2] = v0.z; v[3] = v0.w;
      v[4] = v1.x; v[5] = v1.y; v[6] = v1.z; v[7] = v1.w;
    } else {
#pragma unroll
      for (int i = 0; i < 8; ++i) v[i] = 0.f;     // zero pad == ball origin
    }
    v8h hs, ls;
    float part = 0.f;
#pragma unroll
    for (int i = 0; i < 8; ++i) {
      part += v[i] * v[i];
      const _Float16 h = (_Float16)v[i];
      hs[i] = h;
      ls[i] = (_Float16)(v[i] - (float)h);
    }
    *(v8h*)&sHi[r * kLdsStride + seg * 8] = hs;
    *(v8h*)&sLo[r * kLdsStride + seg * 8] = ls;
    sPart[r * 8 + seg] = part;
  }

#if defined(GYRO_HAVE_TDM)
  if (tid == 0) {
    __builtin_amdgcn_s_wait_tensorcnt(0);         // s_wait_tensorcnt 0
  }
#else
  for (int i = tid; i < kBMatHalfs / 8; i += 256) // fallback cooperative copy
    *(v8h*)&sB[i * 8] = *(const v8h*)(bmat + i * 8);
#endif
  __syncthreads();
  if (tid < 32) {                                 // deterministic |x|^2 per row
    float s = 0.f;
#pragma unroll
    for (int i = 0; i < 8; ++i) s += sPart[tid * 8 + i];
    sX2[tid] = s;
  }
  __syncthreads();

  // ---- WMMA main loop (B operands from LDS) ----
  const int wave  = tid >> 5;
  const int lane  = tid & 31;
  const int rt    = wave >> 2;          // row tile (0/1)
  const int ct0   = (wave & 3) * 2;     // first of two col tiles
  const int lcol  = lane & 15;
  const int lhalf = lane >> 4;
  const int klo   = lhalf * 8;

  v8f accP[2] = {};
  v8f accA[2] = {};

#pragma unroll
  for (int ks = 0; ks < 2; ++ks) {      // K = 0..31, 32..63
    const int kb = ks * 32 + klo;
    const v16h Ah = frag_at(sHi, (rt * 16 + lcol) * kLdsStride + kb);
    const v16h Al = frag_at(sLo, (rt * 16 + lcol) * kLdsStride + kb);
#pragma unroll
    for (int j = 0; j < 2; ++j) {
      const int nOfs = ((ct0 + j) * 16 + lcol) * kD + kb;
      const v16h Bph = frag_at(sB, nOfs);                 // pHi
      const v16h Bpl = frag_at(sB, kO * kD + nOfs);       // pLo
      const v16h Bah = frag_at(sB, 2 * kO * kD + nOfs);   // aHi
      const v16h Bal = frag_at(sB, 3 * kO * kD + nOfs);   // aLo
      accP[j] = wmma_f16(Ah, Bph, accP[j]);
      accP[j] = wmma_f16(Al, Bph, accP[j]);
      accP[j] = wmma_f16(Ah, Bpl, accP[j]);
      accA[j] = wmma_f16(Ah, Bah, accA[j]);
      accA[j] = wmma_f16(Al, Bah, accA[j]);
      accA[j] = wmma_f16(Ah, Bal, accA[j]);
    }
  }

  // ---- Epilogue: Mobius distance + asinh, write dist[m][o] ----
  const int mBase = rowBase + rt * 16;
#pragma unroll
  for (int j = 0; j < 2; ++j) {
    const int o = (ct0 + j) * 16 + lcol;
    const float p2o  = p2s[o];
    const float pao  = pas[o];
    const float ano  = ans[o];
    const float beta = 1.f - p2o;
    union { v8f v; float f[8]; } uP, uA;
    uP.v = accP[j];
    uA.v = accA[j];
#pragma unroll
    for (int e = 0; e < 8; ++e) {
      const int   Mr  = e + lhalf * 8;          // C/D layout: lanes 16-31 -> M+8
      const float x2  = sX2[rt * 16 + Mr];
      const float xdp = uP.f[e];
      const float xda = uA.f[e];
      const float alpha = 1.f - 2.f * xdp + x2;
      const float den   = fmaxf(1.f - 2.f * xdp + p2o * x2, kMinNorm);
      const float sc    = (-alpha * pao + beta * xda) / den;
      const float dn2   = fmaxf((alpha * alpha * p2o - 2.f * alpha * beta * xdp +
                                 beta * beta * x2) / (den * den), kMinNorm);
      const float denom = fmaxf((1.f - dn2) * ano, kMinNorm);
      const float r     = (2.f * sc) / denom;
      dist[(size_t)(mBase + Mr) * kO + o] = asinhf(r);
    }
  }
}

// ---------------------------------------------------------------------------
// Kernel 3: 3x3x3 box filter over padded dist, output (B, O, N, N, N).
// Thread index fastest over x -> coalesced stores; 27 loads hit L0/L2.
// ---------------------------------------------------------------------------
__global__ __launch_bounds__(256) void gyro_box(const float* __restrict__ dist,
                                                float* __restrict__ out) {
  const int idx = blockIdx.x * 256 + threadIdx.x;
  if (idx >= kB * kO * kN * kN * kN) return;
  const int xq = idx % kN;
  int t = idx / kN;
  const int yq = t % kN; t /= kN;
  const int zq = t % kN; t /= kN;
  const int o  = t % kO;
  const int b  = t / kO;
  float s = 0.f;
#pragma unroll
  for (int dz = 0; dz < 3; ++dz)
#pragma unroll
    for (int dy = 0; dy < 3; ++dy)
#pragma unroll
      for (int dx = 0; dx < 3; ++dx) {
        const int m = ((b * kNP + zq + dz) * kNP + (yq + dy)) * kNP + (xq + dx);
        s += dist[(size_t)m * kO + o];
      }
  out[idx] = s;
}

// ---------------------------------------------------------------------------
extern "C" void kernel_launch(void* const* d_in, const int* in_sizes, int n_in,
                              void* d_out, int out_size, void* d_ws, size_t ws_size,
                              hipStream_t stream) {
  const float* x    = (const float*)d_in[0];   // (27000, 16, 64)
  const float* w    = (const float*)d_in[1];   // (128, 64)
  const float* bias = (const float*)d_in[2];   // (128, 1)

  char* ws = (char*)d_ws;
  float* dist = (float*)ws;                                  // 268,435,456 B
  size_t off = kDistElems * sizeof(float);
  _Float16* bmat = (_Float16*)(ws + off);                    // [pHi|pLo|aHi|aLo]
  off += (size_t)kBMatHalfs * sizeof(_Float16);              // 64 KB
  float* p2s = (float*)(ws + off); off += kO * sizeof(float);
  float* pas = (float*)(ws + off); off += kO * sizeof(float);
  float* ans = (float*)(ws + off); off += kO * sizeof(float);

  gyro_prep<<<1, 128, 0, stream>>>(w, bias, bmat, p2s, pas, ans);
  gyro_dist<<<kRows / 32, 256, 0, stream>>>(x, bmat, p2s, pas, ans, dist);
  const int outElems = kB * kO * kN * kN * kN;               // 55,296,000
  gyro_box<<<(outElems + 255) / 256, 256, 0, stream>>>(dist, (float*)d_out);
}
